// MQAAttention_89936615179030
// MI455X (gfx1250) — compile-verified, matrix operand order checked by reading, not run on hardware
//
#include <hip/hip_runtime.h>

typedef unsigned short u16t;
typedef __attribute__((ext_vector_type(16))) __bf16    v16bf;
typedef __attribute__((ext_vector_type(8)))  float     v8f;
typedef __attribute__((ext_vector_type(4)))  unsigned  u32x4;

union FragBf { v16bf v; u32x4 q[2]; };

#define B_  2
#define S_  2048
#define D_  2048
#define H_  16
#define HD_ 128
#define BS_ (B_*S_)

__device__ __forceinline__ u16t f32_to_bf16(float f) {
  union { float f; unsigned u; } c; c.f = f;
  unsigned u = c.u;
  unsigned r = u + 0x7FFFu + ((u >> 16) & 1u);   // round-to-nearest-even
  return (u16t)(r >> 16);
}

__device__ __forceinline__ v8f wmma_bf(v16bf a, v16bf b, v8f c) {
  return __builtin_amdgcn_wmma_f32_16x16x32_bf16(false, a, false, b, (short)0, c,
                                                 false, false);
}

// issue one per-lane async global->LDS 16B copy (tracked by ASYNCcnt)
__device__ __forceinline__ void async_g2l_b128(const u16t* gaddr, u16t* laddr_as3) {
  unsigned long long g = (unsigned long long)gaddr;
  unsigned l = (unsigned)(unsigned long long)laddr_as3;   // LDS byte offset (AS3)
  asm volatile("global_load_async_to_lds_b128 %0, %1, off"
               :: "v"(l), "v"(g) : "memory");
}

// ---------------- elementwise f32 -> bf16 convert ----------------
__global__ __launch_bounds__(256) void conv_bf16_k(const float* __restrict__ in,
                                                   u16t* __restrict__ out, int n) {
  int i = blockIdx.x * 256 + threadIdx.x;
  if (i < n) out[i] = f32_to_bf16(in[i]);
}

// ---------------- bf16 WMMA GEMM: C[M,N] = A[M,K] * W[N,K]^T ----------------
// block = 256 threads (8 waves, 4 M-waves x 2 N-waves); block tile 128x128.
// wave tile: 32(M) x 64(N); K staged through LDS with double-buffered async DMA.
#define LDSROW 40   // 32 bf16 payload padded to 40 (80B) to spread LDS banks
__global__ __launch_bounds__(256) void gemm_bf16_k(const u16t* __restrict__ A,
                                                   const u16t* __restrict__ W,
                                                   float* __restrict__ C,
                                                   int M, int N, int K) {
  __shared__ alignas(16) u16t sA[2][128 * LDSROW];
  __shared__ alignas(16) u16t sB[2][128 * LDSROW];
  int tid  = threadIdx.x;
  int lane = tid & 31;
  int wave = tid >> 5;
  int mw = wave >> 1, nw = wave & 1;
  int mblk = blockIdx.x * 128, nblk = blockIdx.y * 128;
  int m0 = mw * 32;             // local row base of this wave
  int n0 = nw * 64;             // local col base of this wave
  v8f acc[2][4] = {};

  // stage a 128x32 A tile + 128x32 B tile into LDS buffer `buf` (4 async/wave)
  auto stage = [&](int buf, int k0) {
#pragma unroll
    for (int it = 0; it < 2; ++it) {
      int c = tid + it * 256;                 // chunk id 0..511 (16B chunks)
      int row = c >> 2, col16 = c & 3;
      async_g2l_b128(A + (size_t)(mblk + row) * K + k0 + col16 * 8,
                     &sA[buf][row * LDSROW + col16 * 8]);
    }
#pragma unroll
    for (int it = 0; it < 2; ++it) {
      int c = tid + it * 256;
      int row = c >> 2, col16 = c & 3;
      async_g2l_b128(W + (size_t)(nblk + row) * K + k0 + col16 * 8,
                     &sB[buf][row * LDSROW + col16 * 8]);
    }
  };

  stage(0, 0);
  int nk = K >> 5;
  for (int kt = 0; kt < nk; ++kt) {
    int buf = kt & 1;
    if (kt + 1 < nk) {
      stage(buf ^ 1, (kt + 1) << 5);
      asm volatile("s_wait_asynccnt 0x4" ::: "memory");  // current buf landed, next in flight
    } else {
      asm volatile("s_wait_asynccnt 0x0" ::: "memory");
    }
    __syncthreads();                                     // all waves' DMA for buf done

    // materialize all fragments first so ds_loads clause and WMMAs go back-to-back
    FragBf fa[2], fb[4];
    const u16t* ap = &sA[buf][(m0 + (lane & 15)) * LDSROW + ((lane >> 4) << 3)];
    fa[0].q[0] = *(const u32x4*)ap;
    fa[0].q[1] = *(const u32x4*)(ap + 16);
    fa[1].q[0] = *(const u32x4*)(ap + 16 * LDSROW);
    fa[1].q[1] = *(const u32x4*)(ap + 16 * LDSROW + 16);
#pragma unroll
    for (int j = 0; j < 4; ++j) {
      const u16t* bp = &sB[buf][(n0 + j * 16 + (lane & 15)) * LDSROW + ((lane >> 4) << 3)];
      fb[j].q[0] = *(const u32x4*)bp;
      fb[j].q[1] = *(const u32x4*)(bp + 16);
    }
#pragma unroll
    for (int j = 0; j < 4; ++j) {
      acc[0][j] = wmma_bf(fa[0].v, fb[j].v, acc[0][j]);
      acc[1][j] = wmma_bf(fa[1].v, fb[j].v, acc[1][j]);
    }
    __syncthreads();                                     // done reading buf before reuse
  }

#pragma unroll
  for (int i = 0; i < 2; ++i) {
#pragma unroll
    for (int j = 0; j < 4; ++j) {
      float* crow = C + (size_t)(mblk + m0 + i * 16 + 8 * (lane >> 4)) * N
                      + nblk + n0 + j * 16 + (lane & 15);
#pragma unroll
      for (int r = 0; r < 8; ++r) crow[(size_t)r * N] = acc[i][j][r];
    }
  }
}

// ---------------- RoPE + RMSNorm for Q (one wave per (b,s,h) row of 128) ----------------
__global__ __launch_bounds__(256) void rope_norm_q_k(const float* __restrict__ qp,
                                                     u16t* __restrict__ qr,
                                                     const float* __restrict__ w) {
  int lane = threadIdx.x & 31;
  int gid = (blockIdx.x << 3) + (threadIdx.x >> 5);
  int b   = gid >> 15;
  int rem = gid & 32767;
  int s   = rem >> 4;
  int h   = rem & 15;
  const float* row = qp + (size_t)(b * S_ + s) * D_ + h * HD_;
  float o[4];
  float ss = 0.f;
#pragma unroll
  for (int t = 0; t < 2; ++t) {
    int j = lane + t * 32;
    float inv = __expf(-(float)(2 * j) * (9.210340371976184f / (float)HD_));
    float ang = (float)s * inv;
    float c = __cosf(ang), sn = __sinf(ang);
    float x1 = row[j], x2 = row[j + 64];
    float e0 = x1 * c - x2 * sn;
    float e1 = x2 * c + x1 * sn;
    o[t * 2] = e0; o[t * 2 + 1] = e1;
    ss += e0 * e0 + e1 * e1;
  }
  for (int d = 1; d < 32; d <<= 1) ss += __shfl_xor(ss, d, 32);
  float rr = rsqrtf(ss * (1.0f / HD_) + 1e-6f);
  u16t* orow = qr + ((size_t)(b * H_ + h) * S_ + s) * HD_;
#pragma unroll
  for (int t = 0; t < 2; ++t) {
    int j = lane + t * 32;
    orow[j]      = f32_to_bf16(o[t * 2]     * rr * w[j]);
    orow[j + 64] = f32_to_bf16(o[t * 2 + 1] * rr * w[j + 64]);
  }
}

// ---------------- RoPE + RMSNorm for K (one wave per (b,s) row of 128) ----------------
__global__ __launch_bounds__(256) void rope_norm_kk(const float* __restrict__ kp,
                                                    u16t* __restrict__ kr,
                                                    const float* __restrict__ w) {
  int lane = threadIdx.x & 31;
  int gid = (blockIdx.x << 3) + (threadIdx.x >> 5);
  int s = gid & (S_ - 1);
  const float* row = kp + (size_t)gid * HD_;
  float o[4];
  float ss = 0.f;
#pragma unroll
  for (int t = 0; t < 2; ++t) {
    int j = lane + t * 32;
    float inv = __expf(-(float)(2 * j) * (9.210340371976184f / (float)HD_));
    float ang = (float)s * inv;
    float c = __cosf(ang), sn = __sinf(ang);
    float x1 = row[j], x2 = row[j + 64];
    float e0 = x1 * c - x2 * sn;
    float e1 = x2 * c + x1 * sn;
    o[t * 2] = e0; o[t * 2 + 1] = e1;
    ss += e0 * e0 + e1 * e1;
  }
  for (int d = 1; d < 32; d <<= 1) ss += __shfl_xor(ss, d, 32);
  float rr = rsqrtf(ss * (1.0f / HD_) + 1e-6f);
  u16t* orow = kr + (size_t)gid * HD_;
#pragma unroll
  for (int t = 0; t < 2; ++t) {
    int j = lane + t * 32;
    orow[j]      = f32_to_bf16(o[t * 2]     * rr * w[j]);
    orow[j + 64] = f32_to_bf16(o[t * 2 + 1] * rr * w[j + 64]);
  }
}

// ---------------- V transpose + convert: [B*S, HD] f32 -> [B, HD, S] bf16 ----------------
__global__ __launch_bounds__(256) void vtrans_k(const float* __restrict__ vp,
                                                u16t* __restrict__ vt) {
  int i = blockIdx.x * 256 + threadIdx.x;
  int n  = i & (HD_ - 1);
  int bs = i >> 7;
  int b  = bs >> 11;
  int s  = bs & (S_ - 1);
  vt[((size_t)b * HD_ + n) * S_ + s] = f32_to_bf16(vp[i]);
}

// ---------------- causal flash attention ----------------
// grid (S/64, H, B), block 128 (4 waves). wave -> 16 query rows.
__global__ __launch_bounds__(128) void attn_k(const u16t* __restrict__ qr,
                                              const u16t* __restrict__ kr,
                                              const u16t* __restrict__ vt,
                                              float* __restrict__ out) {
  __shared__ alignas(16) u16t pbuf[4][16 * 32];
  int lane = threadIdx.x & 31;
  int wave = threadIdx.x >> 5;
  int b = blockIdx.z, h = blockIdx.y;
  int qbase = blockIdx.x * 64 + wave * 16;
  const u16t* Q  = qr + (size_t)(b * H_ + h) * S_ * HD_;
  const u16t* Kr = kr + (size_t)b * S_ * HD_;
  const u16t* Vt = vt + (size_t)b * HD_ * S_;

  FragBf fq[4];
  const u16t* qrow = Q + (size_t)(qbase + (lane & 15)) * HD_ + ((lane >> 4) << 3);
#pragma unroll
  for (int t = 0; t < 4; ++t) {
    fq[t].q[0] = *(const u32x4*)(qrow + t * 32);
    fq[t].q[1] = *(const u32x4*)(qrow + t * 32 + 16);
  }

  v8f acc[8] = {};
  float rowmax[8], rowsum[8];
#pragma unroll
  for (int r = 0; r < 8; ++r) { rowmax[r] = -1e30f; rowsum[r] = 0.f; }
  const float scale = 0.08838834764831845f;            // 1/sqrt(128)
  int ntiles = ((qbase + 15) >> 5) + 1;                // K-tiles of 32, up to diagonal
  u16t* P = pbuf[wave];

  for (int kt = 0; kt < ntiles; ++kt) {
    int kcol = kt * 32;
    v8f s0 = {}, s1 = {};
    const u16t* krow0 = Kr + (size_t)(kcol + (lane & 15)) * HD_ + ((lane >> 4) << 3);
    const u16t* krow1 = krow0 + 16 * HD_;
#pragma unroll
    for (int t = 0; t < 4; ++t) {
      FragBf fk0, fk1;
      fk0.q[0] = *(const u32x4*)(krow0 + t * 32);
      fk0.q[1] = *(const u32x4*)(krow0 + t * 32 + 16);
      fk1.q[0] = *(const u32x4*)(krow1 + t * 32);
      fk1.q[1] = *(const u32x4*)(krow1 + t * 32 + 16);
      s0 = wmma_bf(fq[t].v, fk0.v, s0);
      s1 = wmma_bf(fq[t].v, fk1.v, s1);
    }
    int rbase = qbase + 8 * (lane >> 4);
#pragma unroll
    for (int r = 0; r < 8; ++r) {
      int row = rbase + r;
      int c0 = kcol + (lane & 15);
      float a0 = (c0      <= row) ? s0[r] * scale : -1e30f;
      float a1 = (c0 + 16 <= row) ? s1[r] * scale : -1e30f;
      float cand = fmaxf(a0, a1);
      for (int d2 = 1; d2 < 16; d2 <<= 1) cand = fmaxf(cand, __shfl_xor(cand, d2, 32));
      float mnew = fmaxf(rowmax[r], cand);
      float sc = __expf(rowmax[r] - mnew);
      rowmax[r] = mnew;
      float p0 = __expf(a0 - mnew), p1 = __expf(a1 - mnew);
      float psum = p0 + p1;
      for (int d2 = 1; d2 < 16; d2 <<= 1) psum += __shfl_xor(psum, d2, 32);
      rowsum[r] = rowsum[r] * sc + psum;
#pragma unroll
      for (int j = 0; j < 8; ++j) acc[j][r] = acc[j][r] * sc;
      int rl = r + 8 * (lane >> 4);
      P[rl * 32 + (lane & 15)]      = f32_to_bf16(p0);
      P[rl * 32 + 16 + (lane & 15)] = f32_to_bf16(p1);
    }
    asm volatile("s_wait_dscnt 0x0" ::: "memory");     // per-wave LDS relayout fence
    FragBf fp;
    const u16t* prow = P + (lane & 15) * 32 + ((lane >> 4) << 3);
    fp.q[0] = *(const u32x4*)(prow);
    fp.q[1] = *(const u32x4*)(prow + 16);
#pragma unroll
    for (int j = 0; j < 8; ++j) {
      FragBf fv;
      const u16t* vp = Vt + (size_t)(j * 16 + (lane & 15)) * S_ + kcol + ((lane >> 4) << 3);
      fv.q[0] = *(const u32x4*)(vp);
      fv.q[1] = *(const u32x4*)(vp + 16);
      acc[j] = wmma_bf(fp.v, fv.v, acc[j]);
    }
  }
#pragma unroll
  for (int j = 0; j < 8; ++j) {
    float* orow = out + ((size_t)b * S_ + qbase + 8 * (lane >> 4)) * D_
                      + h * HD_ + j * 16 + (lane & 15);
#pragma unroll
    for (int r = 0; r < 8; ++r) orow[(size_t)r * D_] = acc[j][r] * (1.0f / rowsum[r]);
  }
}

// ---------------- row RMSNorm over D, f32 in -> bf16 out ----------------
__global__ __launch_bounds__(256) void rmsnorm_rows_k(const float* __restrict__ in,
                                                      const float* __restrict__ w,
                                                      u16t* __restrict__ out, int ncols) {
  __shared__ float red[8];
  int row = blockIdx.x;
  const float* rp = in + (size_t)row * ncols;
  float ss = 0.f;
  for (int c = threadIdx.x; c < ncols; c += 256) { float v = rp[c]; ss += v * v; }
  for (int d = 1; d < 32; d <<= 1) ss += __shfl_xor(ss, d, 32);
  if ((threadIdx.x & 31) == 0) red[threadIdx.x >> 5] = ss;
  __syncthreads();
  float tot = red[0] + red[1] + red[2] + red[3] + red[4] + red[5] + red[6] + red[7];
  float rr = rsqrtf(tot / (float)ncols + 1e-6f);
  u16t* op = out + (size_t)row * ncols;
  for (int c = threadIdx.x; c < ncols; c += 256) op[c] = f32_to_bf16(rp[c] * rr * w[c]);
}

// ---------------- host side ----------------
extern "C" void kernel_launch(void* const* d_in, const int* in_sizes, int n_in,
                              void* d_out, int out_size, void* d_ws, size_t ws_size,
                              hipStream_t stream) {
  const float* x    = (const float*)d_in[0];
  const float* Wq   = (const float*)d_in[1];
  const float* Wk   = (const float*)d_in[2];
  const float* Wv   = (const float*)d_in[3];
  const float* Wo   = (const float*)d_in[4];
  const float* wpre = (const float*)d_in[5];
  const float* wqn  = (const float*)d_in[6];
  const float* wkn  = (const float*)d_in[7];
  float* out = (float*)d_out;
  char* ws = (char*)d_ws;

  const size_t N_X  = (size_t)BS_ * D_;
  const size_t N_WQ = (size_t)D_ * D_;
  const size_t N_WK = (size_t)HD_ * D_;
  const size_t N_KV = (size_t)BS_ * HD_;

  size_t off = 0;
  auto take = [&](size_t bytes) { size_t o = off; off += (bytes + 255) & ~(size_t)255; return o; };
  u16t*  x_bf   = (u16t*)(ws + take(N_X * 2));      // reused as a_norm later
  u16t*  wq_bf  = (u16t*)(ws + take(N_WQ * 2));
  u16t*  wk_bf  = (u16t*)(ws + take(N_WK * 2));
  u16t*  wv_bf  = (u16t*)(ws + take(N_WK * 2));
  u16t*  wo_bf  = (u16t*)(ws + take(N_WQ * 2));
  float* qproj  = (float*)(ws + take(N_X * 4));     // reused as attn_out later
  float* kproj  = (float*)(ws + take(N_KV * 4));
  float* vproj  = (float*)(ws + take(N_KV * 4));
  u16t*  q_rot  = (u16t*)(ws + take(N_X * 2));
  u16t*  k_rot  = (u16t*)(ws + take(N_KV * 2));
  u16t*  v_t    = (u16t*)(ws + take(N_KV * 2));
  float* attn_out = qproj;
  u16t*  a_norm   = x_bf;

  // 1) downconvert operands to bf16
  conv_bf16_k<<<(int)(N_X  / 256), 256, 0, stream>>>(x,  x_bf,  (int)N_X);
  conv_bf16_k<<<(int)(N_WQ / 256), 256, 0, stream>>>(Wq, wq_bf, (int)N_WQ);
  conv_bf16_k<<<(int)(N_WK / 256), 256, 0, stream>>>(Wk, wk_bf, (int)N_WK);
  conv_bf16_k<<<(int)(N_WK / 256), 256, 0, stream>>>(Wv, wv_bf, (int)N_WK);
  conv_bf16_k<<<(int)(N_WQ / 256), 256, 0, stream>>>(Wo, wo_bf, (int)N_WQ);

  // 2) projections (async-LDS staged WMMA GEMM)
  gemm_bf16_k<<<dim3(BS_ / 128, D_ / 128),  256, 0, stream>>>(x_bf, wq_bf, qproj, BS_, D_,  D_);
  gemm_bf16_k<<<dim3(BS_ / 128, HD_ / 128), 256, 0, stream>>>(x_bf, wk_bf, kproj, BS_, HD_, D_);
  gemm_bf16_k<<<dim3(BS_ / 128, HD_ / 128), 256, 0, stream>>>(x_bf, wv_bf, vproj, BS_, HD_, D_);

  // 3) RoPE + RMSNorm, V transpose
  rope_norm_q_k<<<(B_ * S_ * H_) / 8, 256, 0, stream>>>(qproj, q_rot, wqn);
  rope_norm_kk <<<(B_ * S_) / 8,      256, 0, stream>>>(kproj, k_rot, wkn);
  vtrans_k     <<<(int)(N_KV / 256),  256, 0, stream>>>(vproj, v_t);

  // 4) causal flash attention
  attn_k<<<dim3(S_ / 64, H_, B_), 128, 0, stream>>>(q_rot, k_rot, v_t, attn_out);

  // 5) final RMSNorm + output projection
  rmsnorm_rows_k<<<BS_, 256, 0, stream>>>(attn_out, wpre, a_norm, D_);
  gemm_bf16_k<<<dim3(BS_ / 128, D_ / 128), 256, 0, stream>>>(a_norm, wo_bf, out, BS_, D_, D_);
}
